// transformerv1_19189913878581
// MI455X (gfx1250) — matrix operand order repaired and checked by reading, not compiled, not run on hardware
//
#include <hip/hip_runtime.h>
#include <hip/hip_bf16.h>

typedef unsigned short u16;
typedef unsigned int   u32;
typedef __attribute__((ext_vector_type(16))) __bf16 bf16x16;
typedef __attribute__((ext_vector_type(8)))  float  f32x8;
typedef __attribute__((ext_vector_type(8)))  u32    u32x8;
typedef __attribute__((ext_vector_type(4)))  u32    u32x4;

// problem constants
#define TS   128            // sequence length S
#define TD   512            // model dim D
#define TH   8              // heads
#define TDK  64             // head dim
#define TFF  2048           // FFN dim
#define TB   512            // batch
#define TM   (TB * TS)      // 65536 gemm rows
#define PE_COEF (-0.017988946039015984f)   // -ln(10000)/512

__device__ __forceinline__ u16 f2bf(float f) {
  u32 x = __builtin_bit_cast(u32, f);
  u32 r = x + 0x7FFFu + ((x >> 16) & 1u);
  return (u16)(r >> 16);
}

// ---------------------------------------------------------------- convert f32 -> bf16
__global__ __launch_bounds__(256) void tfv1_cvt_kernel(const float* __restrict__ s,
                                                       u16* __restrict__ d, long long n) {
  long long i = (long long)blockIdx.x * 256 + threadIdx.x;
  if (i < n) d[i] = f2bf(s[i]);
}

// convert + transpose weights: src (nMats x K x N) row-major -> dst (nMats x N x K)
__global__ __launch_bounds__(256) void tfv1_cvtT_kernel(const float* __restrict__ s,
                                                        u16* __restrict__ d,
                                                        int Kd, int Nd, long long total) {
  long long i = (long long)blockIdx.x * 256 + threadIdx.x;
  if (i >= total) return;
  long long ms = (long long)Kd * Nd;
  long long mat = i / ms, rem = i - mat * ms;
  int k = (int)(rem / Nd), n = (int)(rem - (long long)k * Nd);
  d[mat * ms + (long long)n * Kd + k] = f2bf(s[i]);
}

// ---------------------------------------------------------------- generic 64x64 WMMA GEMM
// C[m,n] = sum_k A[m,k] * Bt[n,k]  (+bias[n]) (+PE) (+resid[m,n]) (relu)
// MODE: 0 = none, 1 = positional encoding, 2 = relu
template <int MODE, bool HB, bool HR, bool OF, bool OBo>
__global__ __launch_bounds__(128) void tfv1_gemm64_kernel(
    const u16* __restrict__ A, int lda,
    const u16* __restrict__ Bt, int ldbt,   // pre-transposed weights [n][k]
    float* __restrict__ outF, u16* __restrict__ outB, int ldc,
    const float* __restrict__ bias, const float* __restrict__ resid,
    int Kdim) {
  __shared__ u16 As[64 * 32];   // [m][k]
  __shared__ u16 Bs[64 * 32];   // [n][k]

  const int tid  = threadIdx.x;
  const int wave = tid >> 5, lane = tid & 31, half = lane >> 4, r = lane & 15;
  const int m0 = blockIdx.y * 64, n0 = blockIdx.x * 64;

  f32x8 acc[4];
#pragma unroll
  for (int nt = 0; nt < 4; nt++)
#pragma unroll
    for (int i = 0; i < 8; i++) acc[nt][i] = 0.f;

  const int crow = tid >> 1, ckof = (tid & 1) * 16;   // coop-load row + k-offset

  const u16* aSrc = A  + (size_t)(m0 + crow) * lda  + ckof;
  const u16* bSrc = Bt + (size_t)(n0 + crow) * ldbt + ckof;

  for (int k0 = 0; k0 < Kdim; k0 += 32) {
    if (k0 + 32 < Kdim) {   // prefetch next tiles (global_prefetch_b8)
      __builtin_prefetch(aSrc + k0 + 32, 0, 3);
      __builtin_prefetch(bSrc + k0 + 32, 0, 3);
    }
    __syncthreads();
    {  // stage A tile 64x32 and B tile 64x32, both contiguous 32B per thread
      const u32x4* as = (const u32x4*)(aSrc + k0);
      const u32x4* bs = (const u32x4*)(bSrc + k0);
      u32x4* ad = (u32x4*)(As + crow * 32 + ckof);
      u32x4* bd = (u32x4*)(Bs + crow * 32 + ckof);
      ad[0] = as[0]; ad[1] = as[1];
      bd[0] = bs[0]; bd[1] = bs[1];
    }
    __syncthreads();

    // A fragment: lane row = r; element e -> K = half*16+e (contiguous, matches B)
    u32x8 af;
#pragma unroll
    for (int v = 0; v < 8; v++) {
      int kk = half * 16 + 2 * v;
      af[v] = *(const u32*)(As + (wave * 16 + r) * 32 + kk);
    }
#pragma unroll
    for (int nt = 0; nt < 4; nt++) {
      // B fragment: lane col = r, element e -> K = half*16+e (contiguous in Bs[n][k])
      u32x8 bf;
      const u32* bp = (const u32*)(Bs + (nt * 16 + r) * 32 + half * 16);
#pragma unroll
      for (int v = 0; v < 8; v++) bf[v] = bp[v];
      acc[nt] = __builtin_amdgcn_wmma_f32_16x16x32_bf16(
          false, __builtin_bit_cast(bf16x16, af),
          false, __builtin_bit_cast(bf16x16, bf),
          (short)0, acc[nt], false, false);
    }
  }

  // epilogue: C layout element i -> row M = half*8+i, col N = r (per 16x16 tile)
#pragma unroll
  for (int nt = 0; nt < 4; nt++)
#pragma unroll
    for (int i = 0; i < 8; i++) {
      int m = m0 + wave * 16 + half * 8 + i;
      int n = n0 + nt * 16 + r;
      float v = acc[nt][i];
      if (HB) v += bias[n];
      if (MODE == 1) {  // sinusoidal positional encoding
        int s = m & (TS - 1);
        float div = __expf((float)(n & ~1) * PE_COEF);
        float ang = (float)s * div;
        v += (n & 1) ? __cosf(ang) : __sinf(ang);
      }
      size_t off = (size_t)m * ldc + n;
      if (HR) v += resid[off];
      if (MODE == 2) v = fmaxf(v, 0.f);
      if (OF) outF[off] = v;
      if (OBo) outB[off] = f2bf(v);
    }
}

// ---------------------------------------------------------------- fused attention, one WG per (b,h)
__global__ __launch_bounds__(128) void tfv1_attn_kernel(
    const u16* __restrict__ Qg, const u16* __restrict__ Kg,
    const u16* __restrict__ Vg, u16* __restrict__ Og) {
  __shared__ u16 Qs[TS * TDK];        // [t][dk]
  __shared__ u16 Ks[TS * TDK];        // [t][dk]
  __shared__ u16 Vts[TDK * TS];       // transposed: [dv][t]
  __shared__ u16 Ps[4 * 16 * TS];     // per-wave prob strip 16x128, [m][t]

  const int b = blockIdx.x >> 3, h = blockIdx.x & 7;
  const int tid  = threadIdx.x;
  const int wave = tid >> 5, lane = tid & 31, half = lane >> 4, r = lane & 15;

  const size_t gbase = (size_t)b * TS * TD + h * TDK;   // q/k/v/o stored [B*S, H*DK]
  {  // stage Q,K (vectorized) and V transposed (scalar, one-time)
    const u32x4* qs = (const u32x4*)(Qg + gbase + (size_t)tid * TD);
    const u32x4* ks = (const u32x4*)(Kg + gbase + (size_t)tid * TD);
    const u32*   vs = (const u32*)(Vg + gbase + (size_t)tid * TD);
    u32x4* qd = (u32x4*)(Qs + tid * TDK);
    u32x4* kd = (u32x4*)(Ks + tid * TDK);
#pragma unroll
    for (int i = 0; i < 8; i++) { qd[i] = qs[i]; kd[i] = ks[i]; }
#pragma unroll
    for (int v = 0; v < 32; v++) {
      u32 w = vs[v];
      Vts[(2 * v) * TS + tid]     = (u16)(w & 0xFFFFu);
      Vts[(2 * v + 1) * TS + tid] = (u16)(w >> 16);
    }
  }
  __syncthreads();

  for (int iter = 0; iter < 2; iter++) {
    const int rowloc = iter * 64 + wave * 16;   // this wave's 16 query rows

    // ---- scores = (Q K^T) / 8 : 8 key tiles x 2 K-steps of 32
    f32x8 acc[8];
#pragma unroll
    for (int nt = 0; nt < 8; nt++)
#pragma unroll
      for (int i = 0; i < 8; i++) acc[nt][i] = 0.f;

#pragma unroll
    for (int kt = 0; kt < 2; kt++) {
      u32x8 af;
#pragma unroll
      for (int v = 0; v < 8; v++) {
        int kk = kt * 32 + half * 16 + 2 * v;
        af[v] = *(const u32*)(Qs + (rowloc + r) * TDK + kk);
      }
#pragma unroll
      for (int nt = 0; nt < 8; nt++) {
        // B = K^T: lane col = key t = nt*16+r, elements contiguous along dk
        u32x8 bf;
        const u32* bp = (const u32*)(Ks + (nt * 16 + r) * TDK + kt * 32 + half * 16);
#pragma unroll
        for (int v = 0; v < 8; v++) bf[v] = bp[v];
        acc[nt] = __builtin_amdgcn_wmma_f32_16x16x32_bf16(
            false, __builtin_bit_cast(bf16x16, af),
            false, __builtin_bit_cast(bf16x16, bf),
            (short)0, acc[nt], false, false);
      }
    }

    // ---- softmax over 128 cols, 8 rows per lane, cross-lane via shfl_xor in 16-lane half
    f32x8 mx, sum;
#pragma unroll
    for (int i = 0; i < 8; i++) {
      float m = acc[0][i] * 0.125f;
#pragma unroll
      for (int nt = 1; nt < 8; nt++) m = fmaxf(m, acc[nt][i] * 0.125f);
      mx[i] = m; sum[i] = 0.f;
    }
#pragma unroll
    for (int off = 1; off < 16; off <<= 1)
#pragma unroll
      for (int i = 0; i < 8; i++) mx[i] = fmaxf(mx[i], __shfl_xor(mx[i], off, 32));
#pragma unroll
    for (int nt = 0; nt < 8; nt++)
#pragma unroll
      for (int i = 0; i < 8; i++) {
        float e = __expf(acc[nt][i] * 0.125f - mx[i]);
        acc[nt][i] = e; sum[i] += e;
      }
#pragma unroll
    for (int off = 1; off < 16; off <<= 1)
#pragma unroll
      for (int i = 0; i < 8; i++) sum[i] += __shfl_xor(sum[i], off, 32);
    f32x8 inv;
#pragma unroll
    for (int i = 0; i < 8; i++) inv[i] = 1.f / sum[i];

    // ---- store bf16 probs to per-wave LDS strip (row_local = half*8+i, col = nt*16+r)
    u16* pw = Ps + wave * 16 * TS;
#pragma unroll
    for (int nt = 0; nt < 8; nt++)
#pragma unroll
      for (int i = 0; i < 8; i++)
        pw[(half * 8 + i) * TS + nt * 16 + r] = f2bf(acc[nt][i] * inv[i]);
    __syncthreads();

    // ---- O = P @ V : hoisted A-fragments (depend only on kt), V from transposed LDS
    u32x8 pa[4];
#pragma unroll
    for (int kt = 0; kt < 4; kt++)
#pragma unroll
      for (int v = 0; v < 8; v++) {
        int kk = kt * 32 + half * 16 + 2 * v;
        pa[kt][v] = *(const u32*)(pw + r * TS + kk);
      }
#pragma unroll
    for (int vt = 0; vt < 4; vt++) {
      f32x8 a2;
#pragma unroll
      for (int i = 0; i < 8; i++) a2[i] = 0.f;
#pragma unroll
      for (int kt = 0; kt < 4; kt++) {
        // B fragment: lane col dv = vt*16+r, elements contiguous along t in Vts[dv][t]
        u32x8 bf;
        const u32* bp = (const u32*)(Vts + (vt * 16 + r) * TS + kt * 32 + half * 16);
#pragma unroll
        for (int v = 0; v < 8; v++) bf[v] = bp[v];
        a2 = __builtin_amdgcn_wmma_f32_16x16x32_bf16(
            false, __builtin_bit_cast(bf16x16, pa[kt]),
            false, __builtin_bit_cast(bf16x16, bf),
            (short)0, a2, false, false);
      }
#pragma unroll
      for (int i = 0; i < 8; i++) {
        int s = rowloc + half * 8 + i;
        Og[(size_t)(b * TS + s) * TD + h * TDK + vt * 16 + r] = f2bf(a2[i]);
      }
    }
    __syncthreads();
  }
}

// ---------------------------------------------------------------- per-sample mean / rstd over S*D
__global__ __launch_bounds__(256) void tfv1_stats_kernel(const float* __restrict__ rbuf,
                                                         float* __restrict__ stats) {
  const int b = blockIdx.x, t = threadIdx.x;
  const float* p = rbuf + (size_t)b * 65536;
  float s = 0.f, s2 = 0.f;
  for (int i = t; i < 65536; i += 256) { float v = p[i]; s += v; s2 += v * v; }
  __shared__ float sh[256], sh2[256];
  sh[t] = s; sh2[t] = s2; __syncthreads();
  for (int o = 128; o > 0; o >>= 1) {
    if (t < o) { sh[t] += sh[t + o]; sh2[t] += sh2[t + o]; }
    __syncthreads();
  }
  if (t == 0) {
    float mu = sh[0] * (1.f / 65536.f);
    float var = sh2[0] * (1.f / 65536.f) - mu * mu;
    stats[2 * b] = mu;
    stats[2 * b + 1] = rsqrtf(var + 1e-5f);
  }
}

// ---------------------------------------------------------------- layernorm apply -> bf16
__global__ __launch_bounds__(256) void tfv1_norm_kernel(
    const float* __restrict__ rbuf, const float* __restrict__ stats,
    const float* __restrict__ w, const float* __restrict__ bia, u16* __restrict__ rn) {
  size_t idx = (size_t)blockIdx.x * 256 + threadIdx.x;
  int b = (int)(idx >> 16), sd = (int)(idx & 65535);
  float mu = stats[2 * b], rs = stats[2 * b + 1];
  rn[idx] = f2bf((rbuf[idx] - mu) * rs * w[sd] + bia[sd]);
}

// ---------------------------------------------------------------- maxpool 16x16 + fc + log_softmax
__global__ __launch_bounds__(256) void tfv1_poolfc_kernel(
    const float* __restrict__ f, const float* __restrict__ fcw,
    const float* __restrict__ fcb, float* __restrict__ out) {
  const int b = blockIdx.x, t = threadIdx.x;
  const int i = t >> 5, j = t & 31;
  const float* base = f + (size_t)b * 65536 + (size_t)i * 16 * TD + j * 16;
  float m = -3.4e38f;
  for (int rr = 0; rr < 16; rr++)
#pragma unroll
    for (int cc = 0; cc < 16; cc++) m = fmaxf(m, base[rr * TD + cc]);
  __shared__ float s0[256], s1[256];
  s0[t] = m * fcw[2 * t]; s1[t] = m * fcw[2 * t + 1];
  __syncthreads();
  for (int o = 128; o > 0; o >>= 1) {
    if (t < o) { s0[t] += s0[t + o]; s1[t] += s1[t + o]; }
    __syncthreads();
  }
  if (t == 0) {
    float a = s0[0] + fcb[0], c = s1[0] + fcb[1];
    float mx = fmaxf(a, c);
    float lse = mx + __logf(__expf(a - mx) + __expf(c - mx));
    out[2 * b] = a - lse;
    out[2 * b + 1] = c - lse;
  }
}

// ================================================================ host side
extern "C" void kernel_launch(void* const* d_in, const int* in_sizes, int n_in,
                              void* d_out, int out_size, void* d_ws, size_t ws_size,
                              hipStream_t stream) {
  (void)in_sizes; (void)n_in; (void)out_size; (void)ws_size;
  const float* x      = (const float*)d_in[0];
  const float* emb_w  = (const float*)d_in[1];
  const float* emb_b  = (const float*)d_in[2];
  const float* WQ     = (const float*)d_in[3];
  const float* WK     = (const float*)d_in[4];
  const float* WV     = (const float*)d_in[5];
  const float* WO     = (const float*)d_in[6];
  const float* norm_w = (const float*)d_in[7];
  const float* norm_b = (const float*)d_in[8];
  const float* l1_w   = (const float*)d_in[9];
  const float* l1_b   = (const float*)d_in[10];
  const float* l2_w   = (const float*)d_in[11];
  const float* l2_b   = (const float*)d_in[12];
  const float* fc_w   = (const float*)d_in[13];
  const float* fc_b   = (const float*)d_in[14];
  float* out = (float*)d_out;

  // workspace layout (bytes); FF1 overlays dead Q/K/V/O region
  char* W = (char*)d_ws;
  const size_t MB = 1ull << 20;
  float* Af  = (float*)(W);              // 128MB: h fp32 -> r (in place) -> f (ffn out)
  u16* Hb    = (u16*)(W + 128 * MB);     // 64MB  h bf16
  u16* Qb    = (u16*)(W + 192 * MB);     // 64MB  (x bf16 first, then Q)
  u16* Kb    = (u16*)(W + 256 * MB);
  u16* Vb    = (u16*)(W + 320 * MB);
  u16* Ob    = (u16*)(W + 384 * MB);
  u16* FF1   = (u16*)(W + 192 * MB);     // 256MB overlay (Q..O dead by then)
  u16* RN    = (u16*)(W + 448 * MB);     // 64MB
  // pre-transposed bf16 weights [n][k]
  u16* embwT = (u16*)(W + 512 * MB);     // 512x512
  u16* wqT   = (u16*)(W + 513 * MB);     // 8 x 64x512
  u16* wkT   = (u16*)(W + 514 * MB);
  u16* wvT   = (u16*)(W + 515 * MB);
  u16* woT   = (u16*)(W + 516 * MB);     // 512x512
  u16* l1T   = (u16*)(W + 517 * MB);     // 2048x512 (2MB)
  u16* l2T   = (u16*)(W + 519 * MB);     // 512x2048 (2MB)
  float* stats = (float*)(W + 521 * MB);
  u16* xb = Qb;

  // conversions (x plain; weights converted + transposed)
  tfv1_cvt_kernel<<<(TM * TD) / 256, 256, 0, stream>>>(x, xb, (long long)TM * TD);
  auto cvtT = [&](const float* s, u16* d, int Kd, int Nd, int nMats) {
    long long total = (long long)nMats * Kd * Nd;
    tfv1_cvtT_kernel<<<(unsigned)((total + 255) / 256), 256, 0, stream>>>(s, d, Kd, Nd, total);
  };
  cvtT(emb_w, embwT, TD, TD, 1);
  cvtT(WQ, wqT, TD, TDK, TH);
  cvtT(WK, wkT, TD, TDK, TH);
  cvtT(WV, wvT, TD, TDK, TH);
  cvtT(WO, woT, TD, TD, 1);
  cvtT(l1_w, l1T, TD, TFF, 1);
  cvtT(l2_w, l2T, TFF, TD, 1);

  // h = x @ emb_w + emb_b + PE  -> Af (f32) and Hb (bf16)
  tfv1_gemm64_kernel<1, true, false, true, true><<<dim3(TD / 64, TM / 64), 128, 0, stream>>>(
      xb, TD, embwT, TD, Af, Hb, TD, emb_b, nullptr, TD);

  // q,k,v per head into [B*S, H*DK] (col block h*64)
  for (int h = 0; h < TH; h++) {
    dim3 g(1, TM / 64);
    tfv1_gemm64_kernel<0, false, false, false, true><<<g, 128, 0, stream>>>(
        Hb, TD, wqT + h * TD * TDK, TD, nullptr, Qb + h * TDK, TD, nullptr, nullptr, TD);
    tfv1_gemm64_kernel<0, false, false, false, true><<<g, 128, 0, stream>>>(
        Hb, TD, wkT + h * TD * TDK, TD, nullptr, Kb + h * TDK, TD, nullptr, nullptr, TD);
    tfv1_gemm64_kernel<0, false, false, false, true><<<g, 128, 0, stream>>>(
        Hb, TD, wvT + h * TD * TDK, TD, nullptr, Vb + h * TDK, TD, nullptr, nullptr, TD);
  }

  // fused softmax attention
  tfv1_attn_kernel<<<TB * TH, 128, 0, stream>>>(Qb, Kb, Vb, Ob);

  // r = O @ WO + h   (in-place into Af)
  tfv1_gemm64_kernel<0, false, true, true, false><<<dim3(TD / 64, TM / 64), 128, 0, stream>>>(
      Ob, TD, woT, TD, Af, nullptr, TD, nullptr, Af, TD);

  // LayerNorm over (S,D) per sample
  tfv1_stats_kernel<<<TB, 256, 0, stream>>>(Af, stats);
  tfv1_norm_kernel<<<(TB * TS * TD) / 256, 256, 0, stream>>>(Af, stats, norm_w, norm_b, RN);

  // FFN: relu(RN @ l1 + b1) @ l2 + b2 -> Af (f32)
  tfv1_gemm64_kernel<2, true, false, false, true><<<dim3(TFF / 64, TM / 64), 128, 0, stream>>>(
      RN, TD, l1T, TD, nullptr, FF1, TFF, l1_b, nullptr, TD);
  tfv1_gemm64_kernel<0, true, false, true, false><<<dim3(TD / 64, TM / 64), 128, 0, stream>>>(
      FF1, TFF, l2T, TFF, Af, nullptr, TD, l2_b, nullptr, TFF);

  // maxpool + fc + log_softmax
  tfv1_poolfc_kernel<<<TB, 256, 0, stream>>>(Af, fc_w, fc_b, out);
}
